// SIMPT_47270410060247
// MI455X (gfx1250) — compile-verified, matrix operand order checked by reading, not run on hardware
//
#include <hip/hip_runtime.h>
#include <hip/hip_bf16.h>
#include <math.h>

// ---------------------------------------------------------------------------
// Problem constants (fixed by the reference)
// ---------------------------------------------------------------------------
#define Gg 4
#define Nn 4096
#define Tt 512
#define Pp 8
#define Ee 3584                  // edges per graph
#define EG (Gg * Ee)             // 14336 real edges
#define NT (Gg * Nn)             // 16384 nodes total
#define IN_DIM 768
#define HID 256
#define HEADS 4
#define FF 2048
#define OUT_DIM 7
#define HD 64

typedef __attribute__((ext_vector_type(16))) _Float16 v16h;
typedef __attribute__((ext_vector_type(8)))  _Float16 v8h;
typedef __attribute__((ext_vector_type(4)))  _Float16 v4h;
typedef __attribute__((ext_vector_type(8)))  float    v8f;

// ---------------------------------------------------------------------------
// WMMA helpers.  Fragment layouts per CDNA5 ISA 7.12.2 (wave32):
//   A (16x32 f16): lane -> M = lane&15, g = lane>>4; VGPR v holds K pair
//       kb(v,g) = (v<4 ? 2v : 16+2(v-4)) + 8g
//   B (32x16 f16): lane -> N = lane&15; same K mapping
//   C/D (16x16 f32): lane -> N = lane&15; VGPR v -> M = v + 8g
// A and B index patterns are symmetric in (row, k), so a single k-major LDS
// loader serves both: two contiguous 16B runs -> 2x ds_load_b128, no packing.
// ---------------------------------------------------------------------------
__device__ __forceinline__ v8f wmma_f16(v16h a, v16h b, v8f c) {
    return __builtin_amdgcn_wmma_f32_16x16x32_f16(false, a, false, b,
                                                  (short)0, c, false, false);
}

// Fragment from k-major f16 LDS tile: p[row][k], leading dim ld (halves).
// Requires (ld*2) % 16 == 0 for b128 alignment (ld = 40 or 72 both OK).
__device__ __forceinline__ v16h frag_k(const _Float16* p, int ld) {
    int lane = threadIdx.x & 31;
    int r = lane & 15, g = lane >> 4;
    const _Float16* q = p + r * ld + g * 8;
    v8h lo = *(const v8h*)q;          // k = 8g + 0..7   -> f16 pairs v0..v3
    v8h hi = *(const v8h*)(q + 16);   // k = 16+8g + 0..7 -> pairs v4..v7
    return __builtin_shufflevector(lo, hi, 0, 1, 2, 3, 4, 5, 6, 7,
                                   8, 9, 10, 11, 12, 13, 14, 15);
}

__device__ __forceinline__ int frag_kb(int v, int g) {
    return ((v & 3) * 2) + ((v >> 2) * 16) + g * 8;
}

// A fragment straight from global f32 (row-major MxK), scaled, f16-converted
__device__ __forceinline__ v16h frag_a_g(const float* p, int ld, float s) {
    int lane = threadIdx.x & 31;
    int m = lane & 15, g = lane >> 4;
    v16h f;
#pragma unroll
    for (int v = 0; v < 8; ++v) {
        int kb = frag_kb(v, g);
        f[2 * v]     = (_Float16)(p[m * ld + kb] * s);
        f[2 * v + 1] = (_Float16)(p[m * ld + kb + 1] * s);
    }
    return f;
}

__device__ __forceinline__ float red_sum16(float v) {
#pragma unroll
    for (int off = 1; off < 16; off <<= 1) v += __shfl_xor(v, off, 32);
    return v;
}
__device__ __forceinline__ float red_max16(float v) {
#pragma unroll
    for (int off = 1; off < 16; off <<= 1) v = fmaxf(v, __shfl_xor(v, off, 32));
    return v;
}
__device__ __forceinline__ float red_sum32(float v) {
#pragma unroll
    for (int off = 1; off < 32; off <<= 1) v += __shfl_xor(v, off, 32);
    return v;
}

__device__ __forceinline__ v4h cvt4(float4 f) {
    return (v4h){(_Float16)f.x, (_Float16)f.y, (_Float16)f.z, (_Float16)f.w};
}

// ---------------------------------------------------------------------------
// Generic tiled GEMM:  C[M,N] = act( A[M,K] * B + bias + residual )
// A fp32 row-major; B fp32 either [K,N] (transB=0) or [N,K] (transB=1).
// 256 threads / 8 waves, block tile 128x128, wave tile 32x64 (8 WMMAs per
// K-step of 32), double-buffered LDS, b128 staging loads, b64 LDS stores,
// b128 LDS fragment reads.  Last K-step peeled so the hot loop is branchless.
// All call sites have M%128==0, N%128==0, K%32==0.
// ---------------------------------------------------------------------------
#define TM 128
#define TN 128
#define TK 32
#define LDT 40     // padded k-major leading dim (halves): 80B rows, 16B-aligned

__global__ __launch_bounds__(256) void gemm_f16wmma(
    const float* __restrict__ A, const float* __restrict__ B,
    const float* __restrict__ bias, const float* __restrict__ residual,
    float* __restrict__ C, int M, int N, int K, int transB, int relu)
{
    __shared__ _Float16 As[2][TM * LDT];   // [m][k]
    __shared__ _Float16 Bs[2][TN * LDT];   // [n][k]

    int tid = threadIdx.x;
    int bm = blockIdx.y * TM, bn = blockIdx.x * TN;
    int wv = tid >> 5, lane = tid & 31;
    int wm = (wv & 3) * 32;        // wave row offset inside tile
    int wn = (wv >> 2) * 64;       // wave col offset inside tile

    v8f acc[2][4];
#pragma unroll
    for (int i = 0; i < 2; ++i)
#pragma unroll
        for (int j = 0; j < 4; ++j) acc[i][j] = (v8f){};

    float4 areg[4];
    float4 breg[4];
    float  bscal[16];

    auto do_loads = [&](int kk) {
        // A tile: 128 rows x 8 k-quads = 1024 float4s, 4 per thread, coalesced
#pragma unroll
        for (int i = 0; i < 4; ++i) {
            int id = tid + i * 256;
            int m = id >> 3, kq = id & 7;
            areg[i] = *(const float4*)&A[(size_t)(bm + m) * K + kk + kq * 4];
        }
        if (!transB) {
            // B[K,N]: read n-contiguous (coalesced across lanes), 4 k's each
#pragma unroll
            for (int i = 0; i < 4; ++i) {
                int id = tid + i * 256;
                int kq = id >> 7, n = id & 127;
#pragma unroll
                for (int j = 0; j < 4; ++j)
                    bscal[i * 4 + j] = B[(size_t)(kk + kq * 4 + j) * N + bn + n];
            }
        } else {
            // B[N,K]: float4 along k, coalesced across threads
#pragma unroll
            for (int i = 0; i < 4; ++i) {
                int id = tid + i * 256;
                int n = id >> 3, kq = id & 7;
                breg[i] = *(const float4*)&B[(size_t)(bn + n) * K + kk + kq * 4];
            }
        }
    };
    auto do_stores = [&](int buf) {
#pragma unroll
        for (int i = 0; i < 4; ++i) {
            int id = tid + i * 256;
            int m = id >> 3, kq = id & 7;
            *(v4h*)&As[buf][m * LDT + kq * 4] = cvt4(areg[i]);
        }
        if (!transB) {
#pragma unroll
            for (int i = 0; i < 4; ++i) {
                int id = tid + i * 256;
                int kq = id >> 7, n = id & 127;
                v4h h = (v4h){(_Float16)bscal[i * 4 + 0], (_Float16)bscal[i * 4 + 1],
                              (_Float16)bscal[i * 4 + 2], (_Float16)bscal[i * 4 + 3]};
                *(v4h*)&Bs[buf][n * LDT + kq * 4] = h;
            }
        } else {
#pragma unroll
            for (int i = 0; i < 4; ++i) {
                int id = tid + i * 256;
                int n = id >> 3, kq = id & 7;
                *(v4h*)&Bs[buf][n * LDT + kq * 4] = cvt4(breg[i]);
            }
        }
    };
    auto do_mma = [&](int buf) {
        v16h a0 = frag_k(&As[buf][wm * LDT], LDT);
        v16h a1 = frag_k(&As[buf][(wm + 16) * LDT], LDT);
        v16h b0 = frag_k(&Bs[buf][wn * LDT], LDT);
        v16h b1 = frag_k(&Bs[buf][(wn + 16) * LDT], LDT);
        v16h b2 = frag_k(&Bs[buf][(wn + 32) * LDT], LDT);
        v16h b3 = frag_k(&Bs[buf][(wn + 48) * LDT], LDT);
        acc[0][0] = wmma_f16(a0, b0, acc[0][0]);
        acc[0][1] = wmma_f16(a0, b1, acc[0][1]);
        acc[0][2] = wmma_f16(a0, b2, acc[0][2]);
        acc[0][3] = wmma_f16(a0, b3, acc[0][3]);
        acc[1][0] = wmma_f16(a1, b0, acc[1][0]);
        acc[1][1] = wmma_f16(a1, b1, acc[1][1]);
        acc[1][2] = wmma_f16(a1, b2, acc[1][2]);
        acc[1][3] = wmma_f16(a1, b3, acc[1][3]);
    };

    do_loads(0);
    do_stores(0);
    __syncthreads();

    int cur = 0;
    int nsteps = K / TK;
    // Steady state: unconditional loads -> 8 WMMAs -> stores -> barrier
    for (int s = 0; s < nsteps - 1; ++s) {
        int nxt = cur ^ 1;
        do_loads((s + 1) * TK);
        if ((tid & 63) == 0 && (s + 2) < nsteps)
            __builtin_prefetch(&A[(size_t)(bm + (tid >> 1)) * K + (s + 2) * TK], 0, 1);
        do_mma(cur);
        do_stores(nxt);             // nxt buffer not read until after barrier
        __syncthreads();
        cur = nxt;
    }
    do_mma(cur);                    // peeled last step

    int nn = lane & 15, g = lane >> 4;
#pragma unroll
    for (int fi = 0; fi < 2; ++fi)
#pragma unroll
        for (int fj = 0; fj < 4; ++fj)
#pragma unroll
            for (int i = 0; i < 8; ++i) {
                int row = bm + wm + fi * 16 + g * 8 + i;
                int col = bn + wn + fj * 16 + nn;
                float val = acc[fi][fj][i];
                if (bias)     val += bias[col];
                if (residual) val += residual[(size_t)row * N + col];
                if (relu)     val = fmaxf(val, 0.f);
                C[(size_t)row * N + col] = val;
            }
}

// ---------------------------------------------------------------------------
// GAT: per-node attention score dots; also initializes segment max / denom
// ---------------------------------------------------------------------------
__global__ __launch_bounds__(128) void gat_scores(
    const float* __restrict__ xw, const float* __restrict__ a_src,
    const float* __restrict__ a_dst, float* __restrict__ s_src,
    float* __restrict__ s_dst, float* __restrict__ mmax,
    float* __restrict__ denom)
{
    int n = blockIdx.x;
    int h = threadIdx.x >> 5, lane = threadIdx.x & 31;
    const float* xp = xw + (size_t)n * (HEADS * HID) + h * HID;
    const float* as = a_src + h * HID;
    const float* ad = a_dst + h * HID;
    float d0 = 0.f, d1 = 0.f;
#pragma unroll
    for (int i = 0; i < 2; ++i) {
        float4 v = ((const float4*)xp)[lane + i * 32];
        float4 s = ((const float4*)as)[lane + i * 32];
        float4 d = ((const float4*)ad)[lane + i * 32];
        d0 += v.x * s.x + v.y * s.y + v.z * s.z + v.w * s.w;
        d1 += v.x * d.x + v.y * d.y + v.z * d.z + v.w * d.w;
    }
    d0 = red_sum32(d0);
    d1 = red_sum32(d1);
    if (lane == 0) {
        s_src[n * HEADS + h] = d0;
        s_dst[n * HEADS + h] = d1;
        mmax[n * HEADS + h]  = -1e30f;
        denom[n * HEADS + h] = 0.f;
    }
}

__global__ void fill_zero(float* __restrict__ p, int n) {
    int i = blockIdx.x * blockDim.x + threadIdx.x;
    if (i < n) p[i] = 0.f;
}

__device__ __forceinline__ void atomicMaxF(float* a, float v) {
    if (v >= 0.f) atomicMax((int*)a, __float_as_int(v));
    else          atomicMin((unsigned int*)a, (unsigned int)__float_as_int(v));
}

__device__ __forceinline__ void edge_ends(const int* ei, int e, int& s, int& d) {
    if (e < EG) { s = ei[e]; d = ei[EG + e]; }
    else        { s = d = e - EG; }          // self loops appended
}

__global__ void gat_edge_max(const int* __restrict__ ei,
                             const float* __restrict__ s_src,
                             const float* __restrict__ s_dst,
                             float* __restrict__ mmax)
{
    int idx = blockIdx.x * blockDim.x + threadIdx.x;
    if (idx >= (EG + NT) * HEADS) return;
    int e = idx >> 2, h = idx & 3;
    int src, dst;
    edge_ends(ei, e, src, dst);
    float v = s_src[src * HEADS + h] + s_dst[dst * HEADS + h];
    v = v > 0.f ? v : 0.2f * v;              // leaky relu
    atomicMaxF(&mmax[dst * HEADS + h], v);
}

__global__ __launch_bounds__(256) void gat_edge_agg(
    const int* __restrict__ ei, const float* __restrict__ s_src,
    const float* __restrict__ s_dst, const float* __restrict__ mmax,
    const float* __restrict__ xw, float* __restrict__ denom,
    float* __restrict__ agg)
{
    int e = blockIdx.x;
    int src, dst;
    edge_ends(ei, e, src, dst);
    int c = threadIdx.x;
#pragma unroll
    for (int h = 0; h < HEADS; ++h) {
        float v = s_src[src * HEADS + h] + s_dst[dst * HEADS + h];
        v = v > 0.f ? v : 0.2f * v;
        float a = __expf(v - mmax[dst * HEADS + h]);
        if (c == 0) atomicAdd(&denom[dst * HEADS + h], a);
        atomicAdd(&agg[(size_t)dst * (HEADS * HID) + h * HID + c],
                  a * xw[(size_t)src * (HEADS * HID) + h * HID + c]);
    }
}

// mean over heads + bias, then scatter into transformer sequence layout:
// node n = g*N + t*P + p  ->  seq row (g*P + p)*T + t
__global__ __launch_bounds__(256) void gat_finalize(
    const float* __restrict__ agg, const float* __restrict__ denom,
    const float* __restrict__ bias, float* __restrict__ seq)
{
    int n = blockIdx.x, c = threadIdx.x;
    float acc = 0.f;
#pragma unroll
    for (int h = 0; h < HEADS; ++h)
        acc += agg[(size_t)n * (HEADS * HID) + h * HID + c] / denom[n * HEADS + h];
    int g = n >> 12, rem = n & (Nn - 1);
    int t = rem >> 3, p = rem & (Pp - 1);
    seq[((size_t)(g * Pp + p) * Tt + t) * HID + c] = acc * 0.25f + bias[c];
}

// ---------------------------------------------------------------------------
// Flash attention over turns. qkv row layout: [b*T + t][ q(256) k(256) v(256) ].
// Block = 256 threads (8 waves): each wave owns a 16-query tile; the block
// cooperatively stages each 32-key K/V chunk into LDS (vectorized, f16).
// Online softmax entirely in the WMMA C-fragment layout.
// ---------------------------------------------------------------------------
#define LDK 72     // Ks row pitch (halves): [key][d 0..63], 144B rows
#define LDV 40     // Vs row pitch (halves): [d][key 0..31],  80B rows

__global__ __launch_bounds__(256) void attn_flash(
    const float* __restrict__ qkv, float* __restrict__ att_o)
{
    __shared__ _Float16 Ks[32 * LDK];
    __shared__ _Float16 Vs[64 * LDV];
    __shared__ _Float16 Pbuf[8][16 * 40];   // wave-private P transpose bounce

    int tid = threadIdx.x;
    int lane = tid & 31, wv = tid >> 5;
    int b = blockIdx.z, h = blockIdx.y;
    int q0 = (blockIdx.x * 8 + wv) * 16;
    const float* base = qkv + (size_t)b * Tt * (3 * HID);

    // Q fragments with 1/sqrt(64) folded into the f16 conversion (loop-invariant)
    const float* qp = base + (size_t)q0 * (3 * HID) + h * HD;
    v16h qa0 = frag_a_g(qp, 3 * HID, 0.125f);
    v16h qa1 = frag_a_g(qp + 32, 3 * HID, 0.125f);

    float Mr[8], Lr[8];
    v8f o0 = (v8f){}, o1 = (v8f){}, o2 = (v8f){}, o3 = (v8f){};
#pragma unroll
    for (int i = 0; i < 8; ++i) { Mr[i] = -1e30f; Lr[i] = 0.f; }

    for (int kc = 0; kc < Tt; kc += 32) {
        __syncthreads();                     // previous chunk fully consumed
        // Stage K chunk [key][d]: 32x16 float4s, coalesced b128 loads
#pragma unroll
        for (int i = 0; i < 2; ++i) {
            int id = tid + i * 256;
            int key = id >> 4, dq = id & 15;
            float4 kv = *(const float4*)(base + (size_t)(kc + key) * (3 * HID)
                                         + HID + h * HD + dq * 4);
            *(v4h*)&Ks[key * LDK + dq * 4] = cvt4(kv);
        }
        // Stage V chunk transposed [d][key] (scalar, reads coalesced in d)
#pragma unroll
        for (int i = 0; i < 8; ++i) {
            int id = tid + i * 256;
            int d = id & 63, key = id >> 6;
            Vs[d * LDV + key] =
                (_Float16)base[(size_t)(kc + key) * (3 * HID) + 2 * HID + h * HD + d];
        }
        __syncthreads();

        // S = Q * K^T for the 32-key chunk (two 16x16 C fragments)
        v8f s0 = (v8f){}, s1 = (v8f){};
        s0 = wmma_f16(qa0, frag_k(&Ks[0], LDK), s0);
        s0 = wmma_f16(qa1, frag_k(&Ks[32], LDK), s0);
        s1 = wmma_f16(qa0, frag_k(&Ks[16 * LDK], LDK), s1);
        s1 = wmma_f16(qa1, frag_k(&Ks[16 * LDK + 32], LDK), s1);

        // Online softmax (row m = i + 8*(lane>>4), cols across 16 lanes)
#pragma unroll
        for (int i = 0; i < 8; ++i) {
            float r = red_max16(fmaxf(s0[i], s1[i]));
            float mnew = fmaxf(Mr[i], r);
            float al = __expf(Mr[i] - mnew);
            Mr[i] = mnew;
            float e0 = __expf(s0[i] - mnew);
            float e1 = __expf(s1[i] - mnew);
            s0[i] = e0; s1[i] = e1;
            float rs = red_sum16(e0 + e1);
            Lr[i] = Lr[i] * al + rs;
            o0[i] *= al; o1[i] *= al; o2[i] *= al; o3[i] *= al;
        }

        // Transpose P from C-layout to A-layout through wave-private LDS
        int n = lane & 15, g = lane >> 4;
        _Float16* pb = Pbuf[wv];
#pragma unroll
        for (int i = 0; i < 8; ++i) {
            pb[(i + g * 8) * 40 + n]      = (_Float16)s0[i];
            pb[(i + g * 8) * 40 + 16 + n] = (_Float16)s1[i];
        }
        asm volatile("s_wait_dscnt 0" ::: "memory");
        v16h pa = frag_k(pb, 40);

        // O += P(16x32) * V(32x64)
        o0 = wmma_f16(pa, frag_k(&Vs[0], LDV), o0);
        o1 = wmma_f16(pa, frag_k(&Vs[16 * LDV], LDV), o1);
        o2 = wmma_f16(pa, frag_k(&Vs[32 * LDV], LDV), o2);
        o3 = wmma_f16(pa, frag_k(&Vs[48 * LDV], LDV), o3);
    }

    int n = lane & 15, g = lane >> 4;
#pragma unroll
    for (int i = 0; i < 8; ++i) {
        float inv = 1.f / Lr[i];
        int row = b * Tt + q0 + g * 8 + i;
        float* op = att_o + (size_t)row * HID + h * HD;
        op[n]      = o0[i] * inv;
        op[16 + n] = o1[i] * inv;
        op[32 + n] = o2[i] * inv;
        op[48 + n] = o3[i] * inv;
    }
}

// ---------------------------------------------------------------------------
// LayerNorm over HID=256, one wave per row (4 rows per 128-thread block)
// ---------------------------------------------------------------------------
__global__ __launch_bounds__(128) void layernorm_k(
    const float* __restrict__ x, const float* __restrict__ s,
    const float* __restrict__ b, float* __restrict__ y)
{
    int r = blockIdx.x * 4 + (threadIdx.x >> 5);
    int lane = threadIdx.x & 31;
    const float* xp = x + (size_t)r * HID;
    float4 v0 = ((const float4*)xp)[lane];
    float4 v1 = ((const float4*)xp)[lane + 32];
    float sum = v0.x + v0.y + v0.z + v0.w + v1.x + v1.y + v1.z + v1.w;
    sum = red_sum32(sum);
    float mu = sum * (1.f / HID);
    float var = 0.f;
    {
        float d;
        d = v0.x - mu; var += d * d;  d = v0.y - mu; var += d * d;
        d = v0.z - mu; var += d * d;  d = v0.w - mu; var += d * d;
        d = v1.x - mu; var += d * d;  d = v1.y - mu; var += d * d;
        d = v1.z - mu; var += d * d;  d = v1.w - mu; var += d * d;
    }
    var = red_sum32(var) * (1.f / HID);
    float rstd = rsqrtf(var + 1e-5f);
    float4 s0 = ((const float4*)s)[lane],  s1 = ((const float4*)s)[lane + 32];
    float4 b0 = ((const float4*)b)[lane],  b1 = ((const float4*)b)[lane + 32];
    float4 y0, y1;
    y0.x = (v0.x - mu) * rstd * s0.x + b0.x;  y0.y = (v0.y - mu) * rstd * s0.y + b0.y;
    y0.z = (v0.z - mu) * rstd * s0.z + b0.z;  y0.w = (v0.w - mu) * rstd * s0.w + b0.w;
    y1.x = (v1.x - mu) * rstd * s1.x + b1.x;  y1.y = (v1.y - mu) * rstd * s1.y + b1.y;
    y1.z = (v1.z - mu) * rstd * s1.z + b1.z;  y1.w = (v1.w - mu) * rstd * s1.w + b1.w;
    float4* yp = (float4*)(y + (size_t)r * HID);
    yp[lane] = y0;
    yp[lane + 32] = y1;
}

// ---------------------------------------------------------------------------
// GlobalAttention pool over participants + classifier, fused. Block per (g,t).
// ---------------------------------------------------------------------------
__global__ __launch_bounds__(256) void pool_cls(
    const float* __restrict__ seq, const float* __restrict__ pw,
    const float* __restrict__ pb, const float* __restrict__ cw,
    const float* __restrict__ cb, float* __restrict__ out)
{
    __shared__ float logits[Pp];
    __shared__ float pooled[HID];
    int gt = blockIdx.x;
    int g = gt >> 9, t = gt & (Tt - 1);
    int wv = threadIdx.x >> 5, lane = threadIdx.x & 31;

    // gate logits: one wave per participant
    int r = (g * Pp + wv) * Tt + t;
    const float* xp = seq + (size_t)r * HID;
    float d = 0.f;
    for (int c = lane; c < HID; c += 32) d += xp[c] * pw[c];
    d = red_sum32(d);
    if (lane == 0) logits[wv] = d + pb[0];
    __syncthreads();

    // softmax over 8 participants (computed redundantly per thread)
    float mx = -1e30f;
#pragma unroll
    for (int p = 0; p < Pp; ++p) mx = fmaxf(mx, logits[p]);
    float gate[Pp];
    float gsum = 0.f;
#pragma unroll
    for (int p = 0; p < Pp; ++p) { gate[p] = __expf(logits[p] - mx); gsum += gate[p]; }
    float inv = 1.f / gsum;

    int c = threadIdx.x;
    float acc = 0.f;
#pragma unroll
    for (int p = 0; p < Pp; ++p)
        acc += gate[p] * seq[((size_t)(g * Pp + p) * Tt + t) * HID + c];
    pooled[c] = acc * inv;
    __syncthreads();

    // classifier: waves 0..6 each produce one logit
    if (wv < OUT_DIM) {
        float dd = 0.f;
        for (int c2 = lane; c2 < HID; c2 += 32) dd += pooled[c2] * cw[wv * HID + c2];
        dd = red_sum32(dd);
        if (lane == 0) out[(size_t)gt * OUT_DIM + wv] = dd + cb[wv];
    }
}

// ---------------------------------------------------------------------------
// Orchestration
// ---------------------------------------------------------------------------
extern "C" void kernel_launch(void* const* d_in, const int* in_sizes, int n_in,
                              void* d_out, int out_size, void* d_ws, size_t ws_size,
                              hipStream_t stream)
{
    const float* x      = (const float*)d_in[0];
    const int*   ei     = (const int*)d_in[1];
    const float* gat_w  = (const float*)d_in[2];
    const float* a_src  = (const float*)d_in[3];
    const float* a_dst  = (const float*)d_in[4];
    const float* gat_b  = (const float*)d_in[5];
    const float* qkv_w  = (const float*)d_in[6];
    const float* qkv_b  = (const float*)d_in[7];
    const float* out_w  = (const float*)d_in[8];
    const float* out_b  = (const float*)d_in[9];
    const float* ff1_w  = (const float*)d_in[10];
    const float* ff1_b  = (const float*)d_in[11];
    const float* ff2_w  = (const float*)d_in[12];
    const float* ff2_b  = (const float*)d_in[13];
    const float* ln1_s  = (const float*)d_in[14];
    const float* ln1_b  = (const float*)d_in[15];
    const float* ln2_s  = (const float*)d_in[16];
    const float* ln2_b  = (const float*)d_in[17];
    const float* pool_w = (const float*)d_in[18];
    const float* pool_b = (const float*)d_in[19];
    const float* cls_w  = (const float*)d_in[20];
    const float* cls_b  = (const float*)d_in[21];
    float* out = (float*)d_out;

    // Workspace layout (floats); GAT buffers overlapped with transformer ones.
    float* ws   = (float*)d_ws;
    float* bufA = ws;                          // 33.6M: xw (GAT) / ffbuf (FFN)
    float* bufB = bufA + (size_t)16384 * 2048; // 16.8M: agg (GAT) / qkv
    float* seq  = bufB + (size_t)16384 * 1024; // 4.2M
    float* atto = seq  + (size_t)16384 * 256;
    float* t1   = atto + (size_t)16384 * 256;
    float* seq2 = t1   + (size_t)16384 * 256;
    float* ssrc = seq2 + (size_t)16384 * 256;
    float* sdst = ssrc + NT * HEADS;
    float* mmax = sdst + NT * HEADS;
    float* deno = mmax + NT * HEADS;

    // ---- GAT ----
    gemm_f16wmma<<<dim3(1024 / TN, NT / TM), 256, 0, stream>>>(
        x, gat_w, nullptr, nullptr, bufA, NT, HEADS * HID, IN_DIM, 0, 0);
    gat_scores<<<NT, 128, 0, stream>>>(bufA, a_src, a_dst, ssrc, sdst, mmax, deno);
    fill_zero<<<(NT * HEADS * HID + 255) / 256, 256, 0, stream>>>(bufB, NT * HEADS * HID);
    gat_edge_max<<<((EG + NT) * HEADS + 255) / 256, 256, 0, stream>>>(ei, ssrc, sdst, mmax);
    gat_edge_agg<<<EG + NT, 256, 0, stream>>>(ei, ssrc, sdst, mmax, bufA, deno, bufB);
    gat_finalize<<<NT, 256, 0, stream>>>(bufB, deno, gat_b, seq);

    // ---- Transformer encoder, 2 layers ----
    for (int l = 0; l < 2; ++l) {
        gemm_f16wmma<<<dim3(768 / TN, NT / TM), 256, 0, stream>>>(
            seq, qkv_w + (size_t)l * 768 * HID, qkv_b + l * 768,
            nullptr, bufB, NT, 3 * HID, HID, 1, 0);
        attn_flash<<<dim3(Tt / 128, HEADS, Gg * Pp), 256, 0, stream>>>(bufB, atto);
        gemm_f16wmma<<<dim3(HID / TN, NT / TM), 256, 0, stream>>>(
            atto, out_w + (size_t)l * HID * HID, out_b + l * HID,
            seq, t1, NT, HID, HID, 1, 0);
        layernorm_k<<<NT / 4, 128, 0, stream>>>(t1, ln1_s + l * HID, ln1_b + l * HID, seq2);
        gemm_f16wmma<<<dim3(FF / TN, NT / TM), 256, 0, stream>>>(
            seq2, ff1_w + (size_t)l * FF * HID, ff1_b + l * FF,
            nullptr, bufA, NT, FF, HID, 1, 1);
        gemm_f16wmma<<<dim3(HID / TN, NT / TM), 256, 0, stream>>>(
            bufA, ff2_w + (size_t)l * HID * FF, ff2_b + l * HID,
            seq2, t1, NT, HID, FF, 1, 0);
        layernorm_k<<<NT / 4, 128, 0, stream>>>(t1, ln2_s + l * HID, ln2_b + l * HID, seq);
    }

    // ---- Pool + classify ----
    pool_cls<<<Gg * Tt, 256, 0, stream>>>(seq, pool_w, pool_b, cls_w, cls_b, out);
}
